// HomogeneousSchemaGNN_15522011807980
// MI455X (gfx1250) — compile-verified, matrix operand order checked by reading, not run on hardware
//
#include <hip/hip_runtime.h>

// ---------------------------------------------------------------------------
// CDNA5 / gfx1250 GNN forward. Dense GEMMs use v_wmma_f32_16x16x32_bf16
// (f32 accumulate), register-staged double buffering, branch-free predication.
// wave32 everywhere.
// ---------------------------------------------------------------------------

typedef __attribute__((ext_vector_type(16))) __bf16 v16bf;
typedef __attribute__((ext_vector_type(8)))  float  v8f;

#define HID 256
#define BM 64
#define BN 64
#define BK 32
#define LDS_STRIDE (BK + 8)   // bf16 elements; pad to dodge bank conflicts

__device__ __forceinline__ float wave_sum(float s) {
#pragma unroll
    for (int m = 16; m >= 1; m >>= 1) s += __shfl_xor(s, m, 32);
    return s;
}

__device__ __forceinline__ float sigmoidf(float x) {
    return 1.0f / (1.0f + __expf(-x));
}

// ---------------------------------------------------------------------------
// Generic GEMM: C[M,N] = epi( A[M,K] * W[N,K]^T + bias )
// epi: 0 = none, 1 = bias+ReLU
// Requirements (hold at every call site): N % 64 == 0, K % 4 == 0,
// all row strides 16B-aligned.
// ---------------------------------------------------------------------------
__global__ __launch_bounds__(128)
void k_gemm_wmma(const float* __restrict__ A, const float* __restrict__ W,
                 const float* __restrict__ bias, float* __restrict__ C,
                 int M, int N, int K, int epi)
{
    __shared__ __bf16 As[BM][LDS_STRIDE];
    __shared__ __bf16 Bs[BN][LDS_STRIDE];

    const int tid  = threadIdx.x;
    const int lane = tid & 31;
    const int wave = tid >> 5;           // 0..3 -> 16-row strip of the 64-row tile
    const int m0   = blockIdx.x * BM;
    const int n0   = blockIdx.y * BN;

    const v8f vzero = {};
    v8f acc[4];
#pragma unroll
    for (int i = 0; i < 4; ++i) acc[i] = vzero;

    // cooperative-load mapping: 128 threads, one float4 each, 16 rows/pass, 4 passes
    const int lr = tid >> 3;             // 0..15
    const int lc = (tid & 7) * 4;        // 0,4,...,28  (always 4-aligned)

    // register stage buffers (double buffer between global and LDS)
    float4 ra[4], rb[4], na[4], nb[4];

    // ---- prologue: issue loads for tile k0 = 0 (branch-free, clamped addr) ----
    {
        const int  gk  = lc;             // k0 == 0
        const bool inK = gk < K;
#pragma unroll
        for (int p = 0; p < 4; ++p) {
            const int  row = lr + p * 16;
            const int  gm  = m0 + row;
            const bool ok  = (gm < M) & inK;
            const size_t ia = ok  ? ((size_t)gm * K + gk)          : (size_t)0;
            const size_t ib = inK ? ((size_t)(n0 + row) * K + gk)  : (size_t)0;
            ra[p] = *reinterpret_cast<const float4*>(A + ia);
            rb[p] = *reinterpret_cast<const float4*>(W + ib);
        }
    }

    for (int k0 = 0; k0 < K; k0 += BK) {
        const int  gk   = k0 + lc;
        const bool inK  = gk < K;        // K%4==0 && gk%4==0  =>  gk+3 < K too
        const bool more = (k0 + BK) < K;

        // ---- stage registers -> LDS (convert to bf16, zero the out-of-range) ----
#pragma unroll
        for (int p = 0; p < 4; ++p) {
            const int  row = lr + p * 16;
            const bool ok  = ((m0 + row) < M) & inK;
            As[row][lc + 0] = (__bf16)(ok ? ra[p].x : 0.0f);
            As[row][lc + 1] = (__bf16)(ok ? ra[p].y : 0.0f);
            As[row][lc + 2] = (__bf16)(ok ? ra[p].z : 0.0f);
            As[row][lc + 3] = (__bf16)(ok ? ra[p].w : 0.0f);
            Bs[row][lc + 0] = (__bf16)(inK ? rb[p].x : 0.0f);
            Bs[row][lc + 1] = (__bf16)(inK ? rb[p].y : 0.0f);
            Bs[row][lc + 2] = (__bf16)(inK ? rb[p].z : 0.0f);
            Bs[row][lc + 3] = (__bf16)(inK ? rb[p].w : 0.0f);
        }
        __syncthreads();

        // ---- issue global loads for tile k0+BK; they fly during the WMMAs ----
        if (more) {
            const int  gk2  = k0 + BK + lc;
            const bool inK2 = gk2 < K;
#pragma unroll
            for (int p = 0; p < 4; ++p) {
                const int  row = lr + p * 16;
                const int  gm  = m0 + row;
                const bool ok  = (gm < M) & inK2;
                const size_t ia = ok   ? ((size_t)gm * K + gk2)         : (size_t)0;
                const size_t ib = inK2 ? ((size_t)(n0 + row) * K + gk2) : (size_t)0;
                na[p] = *reinterpret_cast<const float4*>(A + ia);
                nb[p] = *reinterpret_cast<const float4*>(W + ib);
            }
        }

        // ---- compute: fragments per ISA 7.12.2 16-bit layouts ----
        // A: lane holds row M = lane%16; lanes<16: K=0..7,16..23; lanes>=16: K=8..15,24..31
        const int mrow = wave * 16 + (lane & 15);
        const int kb   = (lane >> 4) * 8;          // 0 or 8
        v16bf a;
#pragma unroll
        for (int e = 0; e < 8; ++e) {
            a[e]     = As[mrow][kb + e];
            a[e + 8] = As[mrow][kb + 16 + e];
        }
#pragma unroll
        for (int nt = 0; nt < 4; ++nt) {
            const int nrow = nt * 16 + (lane & 15); // B: lane holds col N = lane%16
            v16bf b;
#pragma unroll
            for (int e = 0; e < 8; ++e) {
                b[e]     = Bs[nrow][kb + e];
                b[e + 8] = Bs[nrow][kb + 16 + e];
            }
            acc[nt] = __builtin_amdgcn_wmma_f32_16x16x32_bf16(
                false, a, false, b, (short)0, acc[nt], false, false);
        }
        __syncthreads();

        if (more) {
#pragma unroll
            for (int p = 0; p < 4; ++p) { ra[p] = na[p]; rb[p] = nb[p]; }
        }
    }

    // Epilogue. f32 C/D 16x16 layout: VGPR r holds M=r (lanes 0-15) / M=r+8 (16-31),
    // N = lane%16. N is a multiple of 64 so gn is always in range.
    const int mhalf = (lane >> 4) * 8;
#pragma unroll
    for (int nt = 0; nt < 4; ++nt) {
        const int gn = n0 + nt * 16 + (lane & 15);
        const float bv = (epi != 0) ? bias[gn] : 0.0f;
#pragma unroll
        for (int r = 0; r < 8; ++r) {
            const int gm = m0 + wave * 16 + mhalf + r;
            if (gm < M) {
                float v = acc[nt][r] + bv;
                if (epi != 0) v = v > 0.f ? v : 0.f;
                C[(size_t)gm * N + gn] = v;
            }
        }
    }
}

// ---------------------------------------------------------------------------
// Degree / normalization
// ---------------------------------------------------------------------------
__global__ void k_fill_deg(float* __restrict__ deg, int n) {
    int i = blockIdx.x * blockDim.x + threadIdx.x;
    if (i < n) deg[i] = 1.0f;                 // self-loop contributes 1
}

__global__ void k_deg_edges(float* __restrict__ deg, const long long* __restrict__ dst, int nE) {
    int i = blockIdx.x * blockDim.x + threadIdx.x;
    if (i < nE) atomicAdd(&deg[dst[i]], 1.0f);
}

__global__ void k_dinv(const float* __restrict__ deg, float* __restrict__ dinv, int n) {
    int i = blockIdx.x * blockDim.x + threadIdx.x;
    if (i < n) dinv[i] = rsqrtf(deg[i]);      // deg >= 1 always
}

// agg[n,:] = dinv[n]^2 * xw[n,:]   (self-loop term; also zero-initializes agg)
__global__ void k_selfloop(const float* __restrict__ xw, const float* __restrict__ dinv,
                           float* __restrict__ agg, int total) {
    int i = blockIdx.x * blockDim.x + threadIdx.x;
    if (i < total) {
        int n = i >> 8;                       // / HID
        float d = dinv[n];
        agg[i] = d * d * xw[i];
    }
}

// 64 threads per edge, float4 each; wave covers 512B contiguous in xw (coalesced,
// L2-resident since xw = 51 MB << 192 MB L2).
__global__ void k_scatter(const float* __restrict__ xw, const long long* __restrict__ src,
                          const long long* __restrict__ dst, const float* __restrict__ dinv,
                          float* __restrict__ agg, int nE)
{
    long long tid = (long long)blockIdx.x * blockDim.x + threadIdx.x;
    long long e = tid >> 6;
    if (e >= nE) return;
    const int q = ((int)tid & 63) * 4;
    const long long s = src[e], d = dst[e];
    const float norm = dinv[s] * dinv[d];
    const float4 v = *reinterpret_cast<const float4*>(xw + s * HID + q);
    float* ap = agg + d * HID + q;
    atomicAdd(ap + 0, norm * v.x);
    atomicAdd(ap + 1, norm * v.y);
    atomicAdd(ap + 2, norm * v.z);
    atomicAdd(ap + 3, norm * v.w);
}

// h = LayerNorm(h + agg + bc) * g + b   (in-place; one wave per node, wave32)
__global__ __launch_bounds__(256)
void k_residual_ln(float* __restrict__ h, const float* __restrict__ agg,
                   const float* __restrict__ bc, const float* __restrict__ g,
                   const float* __restrict__ b, int nNodes)
{
    const int wave = threadIdx.x >> 5, lane = threadIdx.x & 31;
    const int node = blockIdx.x * 8 + wave;
    if (node >= nNodes) return;
    const size_t base = (size_t)node * HID;
    float t[8];
    float s = 0.f;
#pragma unroll
    for (int j = 0; j < 8; ++j) {
        const int ch = j * 32 + lane;
        t[j] = h[base + ch] + agg[base + ch] + bc[ch];
        s += t[j];
    }
    const float mu = wave_sum(s) * (1.0f / HID);
    float v = 0.f;
#pragma unroll
    for (int j = 0; j < 8; ++j) { float d0 = t[j] - mu; v += d0 * d0; }
    const float rs = rsqrtf(wave_sum(v) * (1.0f / HID) + 1e-5f);
#pragma unroll
    for (int j = 0; j < 8; ++j) {
        const int ch = j * 32 + lane;
        h[base + ch] = (t[j] - mu) * rs * g[ch] + b[ch];
    }
}

// Per-graph mean+max pool over contiguous sorted batch segments (binary search).
__global__ __launch_bounds__(256)
void k_pool(const float* __restrict__ h, const long long* __restrict__ batch,
            int nNodes, float* __restrict__ gpool)
{
    const int g = blockIdx.x;
    __shared__ int sse[2];
    if (threadIdx.x == 0) {
        for (int which = 0; which < 2; ++which) {
            const long long key = g + which;
            int lo = 0, hi = nNodes;
            while (lo < hi) { int mid = (lo + hi) >> 1; if (batch[mid] < key) lo = mid + 1; else hi = mid; }
            sse[which] = lo;
        }
    }
    __syncthreads();
    const int s = sse[0], e = sse[1];
    const int f = threadIdx.x;  // feature 0..255
    float sum = 0.f, mx = -3.4e38f;
    for (int i = s; i < e; ++i) {
        const float v = h[(size_t)i * HID + f];
        sum += v;
        mx = fmaxf(mx, v);
    }
    const float cnt = (float)(e - s);
    gpool[(size_t)g * (2 * HID) + f]       = sum / cnt;
    gpool[(size_t)g * (2 * HID) + HID + f] = mx;
}

// validity head final layer: logits + sigmoid (one wave per graph)
__global__ __launch_bounds__(256)
void k_validity(const float* __restrict__ vh, const float* __restrict__ Wg2,
                const float* __restrict__ bg2, float* __restrict__ out, int nG)
{
    const int wave = threadIdx.x >> 5, lane = threadIdx.x & 31;
    const int g = blockIdx.x * 8 + wave;
    if (g >= nG) return;
    float s = 0.f;
#pragma unroll
    for (int j = 0; j < 8; ++j) {
        const int ch = j * 32 + lane;
        s += vh[(size_t)g * HID + ch] * Wg2[ch];
    }
    s = wave_sum(s);
    if (lane == 0) {
        const float logit = s + bg2[0];
        out[g]      = sigmoidf(logit);   // [0,64): sigmoid(validity)
        out[nG + g] = logit;             // [64,128): validity logits
    }
}

// node head final layer (one wave per node; nh has 128 features)
__global__ __launch_bounds__(256)
void k_node_logits(const float* __restrict__ nh, const float* __restrict__ Wl2,
                   const float* __restrict__ bl2, float* __restrict__ out,
                   int nNodes, int outBase)
{
    const int wave = threadIdx.x >> 5, lane = threadIdx.x & 31;
    const int n = blockIdx.x * 8 + wave;
    if (n >= nNodes) return;
    float s = 0.f;
#pragma unroll
    for (int j = 0; j < 4; ++j) {
        const int ch = j * 32 + lane;
        s += nh[(size_t)n * 128 + ch] * Wl2[ch];
    }
    s = wave_sum(s);
    if (lane == 0) {
        const float logit = s + bl2[0];
        out[outBase + n]          = sigmoidf(logit);  // sigmoid(node logits)
        out[outBase + nNodes + n] = logit;            // node logits
    }
}

// ---------------------------------------------------------------------------
extern "C" void kernel_launch(void* const* d_in, const int* in_sizes, int n_in,
                              void* d_out, int out_size, void* d_ws, size_t ws_size,
                              hipStream_t stream)
{
    (void)in_sizes; (void)n_in; (void)out_size; (void)ws_size;

    const int N_NODES = 50000, N_EDGES = 1600000, IN_DIM = 404, G = 64, L = 3;

    const float*     x     = (const float*)d_in[0];
    const long long* ei    = (const long long*)d_in[1];
    const long long* batch = (const long long*)d_in[2];
    const float* Win = (const float*)d_in[3];
    const float* bin = (const float*)d_in[4];
    const float* Wc  = (const float*)d_in[5];
    const float* bc  = (const float*)d_in[6];
    const float* lng = (const float*)d_in[7];
    const float* lnb = (const float*)d_in[8];
    const float* Wg1 = (const float*)d_in[9];
    const float* bg1 = (const float*)d_in[10];
    const float* Wg2 = (const float*)d_in[11];
    const float* bg2 = (const float*)d_in[12];
    const float* Wl1 = (const float*)d_in[13];
    const float* bl1 = (const float*)d_in[14];
    const float* Wl2 = (const float*)d_in[15];
    const float* bl2 = (const float*)d_in[16];

    const long long* src = ei;
    const long long* dst = ei + N_EDGES;

    // workspace carve (floats, 64-float aligned chunks)
    float* ws = (float*)d_ws;
    size_t o = 0;
    float* dinv  = ws + o; o += 50048;
    float* deg   = ws + o; o += 50048;
    float* h     = ws + o; o += (size_t)N_NODES * HID;
    float* xw    = ws + o; o += (size_t)N_NODES * HID;
    float* agg   = ws + o; o += (size_t)N_NODES * HID;
    float* gpool = ws + o; o += (size_t)G * 2 * HID;
    float* vh    = ws + o; o += (size_t)G * HID;
    float* nh    = xw;    // reuse (layers done before local head)

    float* out = (float*)d_out;

    // --- degrees / symmetric GCN normalization ---
    k_fill_deg <<<(N_NODES + 255) / 256, 256, 0, stream>>>(deg, N_NODES);
    k_deg_edges<<<(N_EDGES + 255) / 256, 256, 0, stream>>>(deg, dst, N_EDGES);
    k_dinv     <<<(N_NODES + 255) / 256, 256, 0, stream>>>(deg, dinv, N_NODES);

    // --- input projection: h = relu(x @ Win^T + bin) ---
    {
        dim3 grid((N_NODES + BM - 1) / BM, HID / BN);
        k_gemm_wmma<<<grid, 128, 0, stream>>>(x, Win, bin, h, N_NODES, HID, IN_DIM, 1);
    }

    // --- GCN layers ---
    for (int l = 0; l < L; ++l) {
        dim3 grid((N_NODES + BM - 1) / BM, HID / BN);
        k_gemm_wmma<<<grid, 128, 0, stream>>>(h, Wc + (size_t)l * HID * HID, nullptr,
                                              xw, N_NODES, HID, HID, 0);
        const int tot = N_NODES * HID;
        k_selfloop<<<(tot + 255) / 256, 256, 0, stream>>>(xw, dinv, agg, tot);
        const long long work = (long long)N_EDGES * 64;
        k_scatter<<<(unsigned)((work + 255) / 256), 256, 0, stream>>>(xw, src, dst, dinv,
                                                                      agg, N_EDGES);
        k_residual_ln<<<(N_NODES + 7) / 8, 256, 0, stream>>>(h, agg, bc + l * HID,
                                                             lng + l * HID, lnb + l * HID,
                                                             N_NODES);
    }

    // --- pooling + global head ---
    k_pool<<<G, 256, 0, stream>>>(h, batch, N_NODES, gpool);
    {
        dim3 grid((G + BM - 1) / BM, HID / BN);
        k_gemm_wmma<<<grid, 128, 0, stream>>>(gpool, Wg1, bg1, vh, G, HID, 2 * HID, 1);
    }
    k_validity<<<(G + 7) / 8, 256, 0, stream>>>(vh, Wg2, bg2, out, G);

    // --- local head ---
    {
        dim3 grid((N_NODES + BM - 1) / BM, (HID / 2) / BN);
        k_gemm_wmma<<<grid, 128, 0, stream>>>(h, Wl1, bl1, nh, N_NODES, HID / 2, HID, 1);
    }
    k_node_logits<<<(N_NODES + 7) / 8, 256, 0, stream>>>(nh, Wl2, bl2, out, N_NODES, 2 * G);
}
